// FlexAttn_4449586119251
// MI455X (gfx1250) — matrix operand order repaired and checked by reading, not compiled, not run on hardware
//
#include <hip/hip_runtime.h>

typedef __attribute__((ext_vector_type(16))) _Float16 v16h;
typedef __attribute__((ext_vector_type(8)))  _Float16 v8h;
typedef __attribute__((ext_vector_type(4)))  _Float16 v4h;
typedef __attribute__((ext_vector_type(2)))  _Float16 v2h;
typedef __attribute__((ext_vector_type(8)))  float    v8f;

#define L_TOK 680
#define DH    64
#define NQB   11   // ceil(680/64) query blocks of 64 (4 waves x 16)

// Async global->LDS copy, 16B per lane. IOFFSET applies to both the global
// address (VADDR + offset) and the LDS address (LDS_BASE + VDST + offset).
#define ASYNC_CP(ldsoff, gptr, OFFSTR)                                   \
  asm volatile("global_load_async_to_lds_b128 %0, %1, off offset:" OFFSTR \
               :: "v"(ldsoff), "v"(gptr) : "memory")

__global__ __launch_bounds__(128) void var_flash_attn(
    const float* __restrict__ q, const float* __restrict__ k,
    const float* __restrict__ v, float* __restrict__ out)
{
  __shared__ alignas(16) float    lk32[2][32][DH];  // K chunks, f32, double-buffered
  __shared__ alignas(16) float    lv32[2][32][DH];  // V chunks, f32, double-buffered
  __shared__ alignas(16) _Float16 lq[4][16][DH];    // per-wave Q tile, f16, pre-scaled
  __shared__ alignas(16) _Float16 lp[4][16][32];    // per-wave P tile, f16, perm keys

  const int tid  = threadIdx.x;
  const int w    = tid >> 5;       // wave 0..3
  const int lane = tid & 31;
  const int half = lane >> 4;
  const int ln   = lane & 15;

  const int bh  = blockIdx.x / NQB;
  const int qb  = blockIdx.x % NQB;
  const int bq0 = qb * 64;               // block query base
  const int qbase = bq0 + w * 16;        // this wave's query tile base

  const size_t bhoff = (size_t)bh * L_TOK * DH;
  const float* Q = q + bhoff;
  const float* K = k + bhoff;
  const float* V = v + bhoff;
  float* O = out + bhoff;

  const int ends[10] = {1, 5, 14, 30, 55, 91, 155, 255, 424, 680};

  int kendr[8];
#pragma unroll
  for (int r = 0; r < 8; ++r) {
    int qi = qbase + r + 8 * half;
    int e = 680;
    for (int i = 0; i < 10; ++i) { if (qi < ends[i]) { e = ends[i]; break; } }
    kendr[r] = e;
  }
  // this wave's kv extent (0 if the whole tile is past L)
  int lastrow = qbase + 15; if (lastrow >= L_TOK) lastrow = L_TOK - 1;
  int kvmax_own = 680;
  for (int i = 0; i < 10; ++i) { if (lastrow < ends[i]) { kvmax_own = ends[i]; break; } }
  if (qbase >= L_TOK) kvmax_own = 0;
  int kmin = 680;
  for (int i = 0; i < 10; ++i) { if (qbase < ends[i]) { kmin = ends[i]; break; } }
  // block-wide kv extent (from last valid row of the 64-query block)
  int blockLast = bq0 + 63; if (blockLast >= L_TOK) blockLast = L_TOK - 1;
  int kvmax_blk = 680;
  for (int i = 0; i < 10; ++i) { if (blockLast < ends[i]) { kvmax_blk = ends[i]; break; } }

  // per-lane byte offset within each 512B (2-row) block of a 32x64 f32 chunk
  const int laneByte = half * 256 + ln * 16;
  const uint32_t ldskB[2] = {
    (uint32_t)(uintptr_t)&lk32[0][0][0] + (uint32_t)(w * 2048 + laneByte),
    (uint32_t)(uintptr_t)&lk32[1][0][0] + (uint32_t)(w * 2048 + laneByte) };
  const uint32_t ldsvB[2] = {
    (uint32_t)(uintptr_t)&lv32[0][0][0] + (uint32_t)(w * 2048 + laneByte),
    (uint32_t)(uintptr_t)&lv32[1][0][0] + (uint32_t)(w * 2048 + laneByte) };
  const char* Kc = (const char*)K;
  const char* Vc = (const char*)V;

  // ---- stage this wave's Q tile (scaled by 1/8), batched loads ----
#pragma unroll
  for (int g = 0; g < 2; ++g) {
    float4 f[4];
#pragma unroll
    for (int u = 0; u < 4; ++u) {
      int row = 2 * (4 * g + u) + half;
      int grow = qbase + row; if (grow >= L_TOK) grow = L_TOK - 1;
      f[u] = *(const float4*)(Q + (size_t)grow * DH + ln * 4);
    }
#pragma unroll
    for (int u = 0; u < 4; ++u) {
      int row = 2 * (4 * g + u) + half;
      v4h hv;
      hv[0] = (_Float16)(f[u].x * 0.125f); hv[1] = (_Float16)(f[u].y * 0.125f);
      hv[2] = (_Float16)(f[u].z * 0.125f); hv[3] = (_Float16)(f[u].w * 0.125f);
      *(v4h*)&lq[w][row][ln * 4] = hv;
    }
  }

  // each wave stages its 1/4 of a K/V chunk: rows 8w..8w+7 (4 x 512B each)
  auto issue_share = [&](int ci, int buf) {
    int kv0 = ci << 5;
    uint32_t ldk = ldskB[buf];
    uint32_t ldv = ldsvB[buf];
    if (kv0 + 32 <= L_TOK) {
      const char* gk = Kc + (size_t)kv0 * 256 + (size_t)(w * 2048 + laneByte);
      const char* gv = Vc + (size_t)kv0 * 256 + (size_t)(w * 2048 + laneByte);
      ASYNC_CP(ldk, gk, "0");   ASYNC_CP(ldk, gk, "512");
      ASYNC_CP(ldk, gk, "1024"); ASYNC_CP(ldk, gk, "1536");
      ASYNC_CP(ldv, gv, "0");   ASYNC_CP(ldv, gv, "512");
      ASYNC_CP(ldv, gv, "1024"); ASYNC_CP(ldv, gv, "1536");
    } else {
#pragma unroll
      for (int t = 0; t < 4; ++t) {
        int r = kv0 + 8 * w + 2 * t + half; if (r >= L_TOK) r = L_TOK - 1;
        const char* gk = Kc + (size_t)r * 256 + (size_t)(ln * 16);
        const char* gv = Vc + (size_t)r * 256 + (size_t)(ln * 16);
        ASYNC_CP(ldk + (uint32_t)(t * 512), gk, "0");
        ASYNC_CP(ldv + (uint32_t)(t * 512), gv, "0");
      }
    }
  };

  v16h aq0, aq1;
  v16h bones;
  {
    _Float16 one = (_Float16)((ln == 0) ? 1.0f : 0.0f);
#pragma unroll
    for (int j = 0; j < 16; ++j) bones[j] = one;
  }

  v8f oacc[4];
  v8f osum;
#pragma unroll
  for (int c = 0; c < 4; ++c)
#pragma unroll
    for (int j = 0; j < 8; ++j) oacc[c][j] = 0.0f;
#pragma unroll
  for (int j = 0; j < 8; ++j) osum[j] = 0.0f;

  const int nch = (kvmax_blk + 31) >> 5;
  issue_share(0, 0);                       // prologue DMA
  asm volatile("s_wait_asynccnt 0x0" ::: "memory");
  __syncthreads();                         // Q tiles + chunk 0 visible

  // A-fragments for Q (after barrier)
  {
    int off = half * 16;
    v8h l0 = *(const v8h*)&lq[w][ln][ 0 + off];
    v8h h0 = *(const v8h*)&lq[w][ln][ 8 + off];
    v8h l1 = *(const v8h*)&lq[w][ln][32 + off];
    v8h h1 = *(const v8h*)&lq[w][ln][40 + off];
#pragma unroll
    for (int j = 0; j < 8; ++j) {
      aq0[j] = l0[j]; aq0[j + 8] = h0[j];
      aq1[j] = l1[j]; aq1[j + 8] = h1[j];
    }
  }

  for (int ci = 0; ci < nch; ++ci) {
    const int kv0 = ci << 5;
    const int cur = ci & 1;
    const bool more = (ci + 1 < nch);
    if (more) issue_share(ci + 1, cur ^ 1);   // overlap next chunk DMA

    if (kv0 < kvmax_own) {
      // --- S = (Q*scale) @ K^T : 2 tiles x 2 WMMAs, K read f32 -> f16 ---
      v8f s[2];
#pragma unroll
      for (int t = 0; t < 2; ++t) {
        int kr = 16 * t + ln;
        const float* s0p = &lk32[cur][kr][half * 16];
        const float* s1p = &lk32[cur][kr][32 + half * 16];
        float4 w0 = ((const float4*)s0p)[0], w1 = ((const float4*)s0p)[1];
        float4 w2 = ((const float4*)s0p)[2], w3 = ((const float4*)s0p)[3];
        float4 x0 = ((const float4*)s1p)[0], x1 = ((const float4*)s1p)[1];
        float4 x2 = ((const float4*)s1p)[2], x3 = ((const float4*)s1p)[3];
        v16h bk0, bk1;
        bk0[0]=(_Float16)w0.x; bk0[1]=(_Float16)w0.y; bk0[2]=(_Float16)w0.z; bk0[3]=(_Float16)w0.w;
        bk0[4]=(_Float16)w1.x; bk0[5]=(_Float16)w1.y; bk0[6]=(_Float16)w1.z; bk0[7]=(_Float16)w1.w;
        bk0[8]=(_Float16)w2.x; bk0[9]=(_Float16)w2.y; bk0[10]=(_Float16)w2.z; bk0[11]=(_Float16)w2.w;
        bk0[12]=(_Float16)w3.x; bk0[13]=(_Float16)w3.y; bk0[14]=(_Float16)w3.z; bk0[15]=(_Float16)w3.w;
        bk1[0]=(_Float16)x0.x; bk1[1]=(_Float16)x0.y; bk1[2]=(_Float16)x0.z; bk1[3]=(_Float16)x0.w;
        bk1[4]=(_Float16)x1.x; bk1[5]=(_Float16)x1.y; bk1[6]=(_Float16)x1.z; bk1[7]=(_Float16)x1.w;
        bk1[8]=(_Float16)x2.x; bk1[9]=(_Float16)x2.y; bk1[10]=(_Float16)x2.z; bk1[11]=(_Float16)x2.w;
        bk1[12]=(_Float16)x3.x; bk1[13]=(_Float16)x3.y; bk1[14]=(_Float16)x3.z; bk1[15]=(_Float16)x3.w;
        v8f acc = {0.f, 0.f, 0.f, 0.f, 0.f, 0.f, 0.f, 0.f};
        acc = __builtin_amdgcn_wmma_f32_16x16x32_f16(false, aq0, false, bk0,
                                                     (short)0, acc, false, false);
        acc = __builtin_amdgcn_wmma_f32_16x16x32_f16(false, aq1, false, bk1,
                                                     (short)0, acc, false, false);
        s[t] = acc;
      }

      // --- P = exp(S); masked -> 0; packed pairs in permuted key order ---
      if (kv0 + 32 <= kmin) {
#pragma unroll
        for (int r = 0; r < 8; ++r) {
          int row = r + 8 * half;
          v2h pe;
          pe[0] = (_Float16)__expf(s[0][r]);
          pe[1] = (_Float16)__expf(s[1][r]);
          *(v2h*)&lp[w][row][2 * ln] = pe;
        }
      } else {
        int j0 = kv0 + ln;
        int j1 = kv0 + 16 + ln;
#pragma unroll
        for (int r = 0; r < 8; ++r) {
          int row = r + 8 * half;
          v2h pe;
          pe[0] = (_Float16)((j0 < kendr[r]) ? __expf(s[0][r]) : 0.0f);
          pe[1] = (_Float16)((j1 < kendr[r]) ? __expf(s[1][r]) : 0.0f);
          *(v2h*)&lp[w][row][2 * ln] = pe;
        }
      }

      // A fragment of P (contiguous reads in permuted-kk space)
      v16h pa;
      {
        int off = half * 16;
        v8h lo = *(const v8h*)&lp[w][ln][ 0 + off];
        v8h hi = *(const v8h*)&lp[w][ln][ 8 + off];
#pragma unroll
        for (int j = 0; j < 8; ++j) { pa[j] = lo[j]; pa[j + 8] = hi[j]; }
      }

      // --- O += P @ V : key(j) = (j&1)*16 + (j>>1) + 8*half (perm gather) ---
#pragma unroll
      for (int c = 0; c < 4; ++c) {
        int col = 16 * c + ln;
        v16h bv;
#pragma unroll
        for (int j = 0; j < 16; ++j) {
          int key = ((j & 1) << 4) + (j >> 1) + 8 * half;
          bv[j] = (_Float16)lv32[cur][key][col];
        }
        oacc[c] = __builtin_amdgcn_wmma_f32_16x16x32_f16(false, pa, false, bv,
                                                         (short)0, oacc[c], false, false);
      }
      osum = __builtin_amdgcn_wmma_f32_16x16x32_f16(false, pa, false, bones,
                                                    (short)0, osum, false, false);
    }

    if (more) asm volatile("s_wait_asynccnt 0x0" ::: "memory"); // own share of ci+1
    __syncthreads();   // publish DMA shares; retire all readers of buf[cur]
  }

  // ---- normalize and store ----
#pragma unroll
  for (int r = 0; r < 8; ++r) {
    float ssum = __shfl(osum[r], 0, 16);
    int qi = qbase + r + 8 * half;
    if (qi < L_TOK) {
      float inv = 1.0f / ssum;
#pragma unroll
      for (int c = 0; c < 4; ++c)
        O[(size_t)qi * DH + 16 * c + ln] = oacc[c][r] * inv;
    }
  }
}

extern "C" void kernel_launch(void* const* d_in, const int* in_sizes, int n_in,
                              void* d_out, int out_size, void* d_ws, size_t ws_size,
                              hipStream_t stream) {
  const float* q = (const float*)d_in[0];
  const float* k = (const float*)d_in[1];
  const float* v = (const float*)d_in[2];
  float* out = (float*)d_out;
  const int B = 8, H = 16;
  dim3 grid(B * H * NQB);
  dim3 block(128);
  var_flash_attn<<<grid, block, 0, stream>>>(q, k, v, out);
}
